// MambaForTokenClassification_83305185673317
// MI455X (gfx1250) — matrix operand order repaired and checked by reading, not compiled, not run on hardware
//
#include <hip/hip_runtime.h>
#include <hip/hip_bf16.h>

// ---------------------------------------------------------------------------
// Mamba token-classification forward for MI455X (gfx1250, wave32, WMMA).
//   - bf16 WMMA (v_wmma_f32_16x16x32_bf16) for all M=4096 GEMMs
//   - GLOBAL_LOAD_ASYNC_TO_LDS_B128 staging (ASYNCcnt) in the GEMM
//   - Tensor Data Mover (tensor_load_to_lds + s_wait_tensorcnt) in the scan
// ---------------------------------------------------------------------------

typedef __bf16 bf16;
typedef bf16  v8bf  __attribute__((ext_vector_type(8)));
typedef bf16  v16bf __attribute__((ext_vector_type(16)));
typedef float v8f   __attribute__((ext_vector_type(8)));

typedef unsigned int u32x4 __attribute__((ext_vector_type(4)));
typedef int          i32x8 __attribute__((ext_vector_type(8)));
typedef int          i32x4 __attribute__((ext_vector_type(4)));

constexpr int D_IN = 768, D_M = 1536, D_S = 16, D_D = 96, K_CONV = 4;
constexpr int N_LAB = 9, N_LAYERS = 2, BB = 2, SS = 2048;
constexpr int MROWS = BB * SS;   // 4096 token rows

__device__ __forceinline__ bf16  f2bf(float x) { return (bf16)x; }
__device__ __forceinline__ float sigf(float x) { return 1.f / (1.f + __expf(-x)); }

// Flat address of a __shared__ object -> wave-logical LDS byte offset.
// (ISA 10.2: LDS aperture addresses map to LDS via ADDR[31:0].)
__device__ __forceinline__ unsigned lds_off(const void* p) {
    return (unsigned)(unsigned long long)p;
}

// ---------------------------------------------------------------------------
// TDM: 1-row tile of `nelem` 4-byte elements, global -> LDS (ISA ch.8 D#).
// Issued once per wave; caller guards to a single wave and waits TENSORcnt.
// 6-arg builtin form (clang-23 / therock-10.0 headers).
// ---------------------------------------------------------------------------
__device__ __forceinline__ void tdm_load_lds_1d(unsigned lds_addr, const void* gptr,
                                                unsigned nelem)
{
    const unsigned long long ga = (unsigned long long)gptr;
    const unsigned ga_lo = __builtin_amdgcn_readfirstlane((unsigned)(ga & 0xFFFFFFFFull));
    const unsigned ga_hi = __builtin_amdgcn_readfirstlane((unsigned)(ga >> 32));
    const unsigned laddr = __builtin_amdgcn_readfirstlane(lds_addr);

    u32x4 g0;
    g0[0] = 1u;                                   // count=1, user descriptor
    g0[1] = laddr;                                // lds_addr
    g0[2] = ga_lo;                                // global_addr[31:0]
    g0[3] = (ga_hi & 0x1FFFFFFu) | (2u << 30);    // global_addr[56:32] | type=2

    i32x8 g1;
    g1[0] = (int)(2u << 16);                      // data_size = 4B; no pad/iterate
    g1[1] = (int)(nelem << 16);                   // tensor_dim0[15:0]
    g1[2] = (int)((nelem >> 16) | (1u << 16));    // tensor_dim0[31:16] | tensor_dim1=1
    g1[3] = (int)(nelem << 16);                   // tile_dim0 = nelem
    g1[4] = 1;                                    // tile_dim1 = 1
    g1[5] = (int)nelem;                           // tensor_dim0_stride = nelem
    g1[6] = 0;
    g1[7] = 0;

    i32x4 z4 = {0, 0, 0, 0};
    i32x8 z8 = {0, 0, 0, 0, 0, 0, 0, 0};
    __builtin_amdgcn_tensor_load_to_lds(g0, g1, z4, z4, z8, 0);
}

// ---------------------------------------------------------------------------
// WMMA NT-GEMM: C[M,N] (f32) = A[M,K] (bf16 row-major) * W[N,K]^T (bf16 row-major)
// Workgroup: 256 threads = 8 waves; tile 128(M) x 64(N); K-step 32.
// Staging via GLOBAL_LOAD_ASYNC_TO_LDS_B128 (no VGPR round-trip).
// ---------------------------------------------------------------------------
#define BM 128
#define BN 64
#define BK 32
#define LDSTR 40   // halves per LDS row: 32 data + 8 pad -> 80B stride, 16B aligned

__global__ __launch_bounds__(256) void wmma_gemm_nt(
    const bf16* __restrict__ A, const bf16* __restrict__ W,
    float* __restrict__ C, int M, int N, int K)
{
    __shared__ __align__(16) bf16 sA[BM * LDSTR];
    __shared__ __align__(16) bf16 sB[BN * LDSTR];

    const int tid  = threadIdx.x;
    const int wave = tid >> 5;
    const int lane = tid & 31;
    const int m0   = blockIdx.y * BM;
    const int n0   = blockIdx.x * BN;

    v8f acc[4] = {};

    // global->LDS staging map
    const int arow = tid >> 1;            // 128 rows of A, 32B (2x b128) per thread
    const int aseg = (tid & 1) * 16;
    const int brow = tid >> 2;            // 64 rows of W, 16B per thread
    const int bseg = (tid & 3) * 8;
    // Clamp OOB weight rows: garbage only reaches N-tiles that are never stored.
    const int nclamp = (n0 + brow < N) ? (n0 + brow) : (N - 1);

    const unsigned long long agp =
        (unsigned long long)(A + (size_t)(m0 + arow) * K + aseg);
    const unsigned long long bgp =
        (unsigned long long)(W + (size_t)nclamp * K + bseg);
    const unsigned lA = lds_off(&sA[arow * LDSTR + aseg]);
    const unsigned lB = lds_off(&sB[brow * LDSTR + bseg]);

    const int khalf = lane >> 4;
    const int lrow  = lane & 15;

    for (int k0 = 0; k0 < K; k0 += BK) {
        const unsigned long long ga = agp + (unsigned long long)k0 * 2u;
        const unsigned long long gb = bgp + (unsigned long long)k0 * 2u;
        // async copy: 32B of A + 16B of W per thread, straight into LDS
        asm volatile(
            "global_load_async_to_lds_b128 %0, %2, off\n\t"
            "global_load_async_to_lds_b128 %0, %2, off offset:16\n\t"
            "global_load_async_to_lds_b128 %1, %3, off"
            :: "v"(lA), "v"(lB), "v"(ga), "v"(gb)
            : "memory");
        if (k0 + BK < K)
            __builtin_prefetch((const void*)(ga + 2 * BK), 0, 1);
        asm volatile("s_wait_asynccnt 0" ::: "memory");
        __syncthreads();

        // ---- A fragment (lane m = lane&15; K in {8kh..+8} U {16+8kh..+8}) ----
        const bf16* ap = &sA[(wave * 16 + lrow) * LDSTR];
        union { v16bf full; v8bf h[2]; } fa;
        fa.h[0] = *(const v8bf*)(ap + khalf * 8);
        fa.h[1] = *(const v8bf*)(ap + 16 + khalf * 8);

        // ---- 4 N-tiles: B fragment lane n holds K in {8kh..+8} U {16+8kh..+8} ----
        #pragma unroll
        for (int nt = 0; nt < 4; ++nt) {
            const bf16* bp = &sB[(nt * 16 + lrow) * LDSTR];
            union { v16bf full; v8bf h[2]; } fb;
            fb.h[0] = *(const v8bf*)(bp + khalf * 8);
            fb.h[1] = *(const v8bf*)(bp + 16 + khalf * 8);
            acc[nt] = __builtin_amdgcn_wmma_f32_16x16x32_bf16(
                false, fa.full, false, fb.full, (short)0, acc[nt], false, false);
        }
        __syncthreads();
    }

    // ---- store: VGPR r -> M = r + 8*(lane>=16), N = lane&15 ----
    const int mrow = m0 + wave * 16 + ((lane >> 4) << 3);
    #pragma unroll
    for (int nt = 0; nt < 4; ++nt) {
        const int n = n0 + nt * 16 + lrow;
        if (n < N) {
            float* cp = C + (size_t)mrow * N + n;
            #pragma unroll
            for (int r = 0; r < 8; ++r)
                cp[(size_t)r * N] = acc[nt][r];
        }
    }
}

// ---------------------------------------------------------------------------
// Elementwise / small kernels
// ---------------------------------------------------------------------------
__global__ void cvt_bf16_kernel(const float* __restrict__ in, bf16* __restrict__ out, size_t n)
{
    size_t i = (size_t)blockIdx.x * blockDim.x + threadIdx.x;
    if (i < n) out[i] = f2bf(in[i]);
}

__global__ __launch_bounds__(256) void embed_kernel(
    const int* __restrict__ ids, const int* __restrict__ amask,
    const float* __restrict__ emb, float* __restrict__ x, int dim)
{
    const int row = blockIdx.x;
    const int id  = ids[row];
    const float mk = (float)amask[row];
    const float* e = emb + (size_t)id * dim;
    float* xr = x + (size_t)row * dim;
    for (int k = threadIdx.x; k < dim; k += 256) xr[k] = e[k] * mk;
}

__global__ __launch_bounds__(256) void rmsnorm_bf16_kernel(
    const float* __restrict__ x, const float* __restrict__ w,
    bf16* __restrict__ out, int dim)
{
    __shared__ float red[8];
    const int row = blockIdx.x;
    const float* xr = x + (size_t)row * dim;
    float ss = 0.f;
    for (int k = threadIdx.x; k < dim; k += 256) { float v = xr[k]; ss += v * v; }
    for (int off = 16; off > 0; off >>= 1) ss += __shfl_down(ss, off, 32);
    if ((threadIdx.x & 31) == 0) red[threadIdx.x >> 5] = ss;
    __syncthreads();
    if (threadIdx.x == 0) {
        float t = 0.f;
        for (int i = 0; i < 8; ++i) t += red[i];
        red[0] = rsqrtf(t / dim + 1e-6f);
    }
    __syncthreads();
    const float scale = red[0];
    bf16* orow = out + (size_t)row * dim;
    for (int k = threadIdx.x; k < dim; k += 256) orow[k] = f2bf(xr[k] * scale * w[k]);
}

// causal depthwise conv (K=4) + bias + SiLU over the 'a' half of ag
__global__ __launch_bounds__(256) void conv_silu_kernel(
    const float* __restrict__ ag, const float* __restrict__ cw,
    const float* __restrict__ cb, float* __restrict__ af, bf16* __restrict__ abf,
    int L, int dm)
{
    size_t idx = (size_t)blockIdx.x * 256 + threadIdx.x;
    size_t total = (size_t)MROWS * dm;
    if (idx >= total) return;
    const int d = (int)(idx % dm);
    const size_t bl = idx / dm;
    const int l = (int)(bl % L);
    const int b = (int)(bl / L);
    const float* arow = ag + (size_t)b * L * (2 * dm) + d;  // ag[b, :, d]
    float acc = cb[d];
    #pragma unroll
    for (int i = 0; i < K_CONV; ++i) {
        const int ls = l - (K_CONV - 1) + i;
        if (ls >= 0) acc += arow[(size_t)ls * (2 * dm)] * cw[d * K_CONV + i];
    }
    const float s = acc * sigf(acc);
    af[idx] = s;
    abf[idx] = f2bf(s);
}

__global__ void softplus_kernel(float* __restrict__ delta, const float* __restrict__ Dp,
                                size_t n, int dm)
{
    size_t i = (size_t)blockIdx.x * blockDim.x + threadIdx.x;
    if (i >= n) return;
    const float v = Dp[i % dm] + delta[i];
    delta[i] = (v > 20.f) ? v : log1pf(__expf(v));
}

// selective scan: thread = (batch, channel); 16 states in registers;
// B/C chunks brought into LDS by the Tensor Data Mover (wave 0).
#define SCHUNK 32
__global__ __launch_bounds__(256) void scan_kernel(
    const float* __restrict__ delta, const float* __restrict__ af,
    const float* __restrict__ Bm, const float* __restrict__ Cm,
    const float* __restrict__ Amat, const float* __restrict__ Dp,
    float* __restrict__ y, int L, int dm)
{
    __shared__ __align__(16) float sB[SCHUNK * D_S];
    __shared__ __align__(16) float sC[SCHUNK * D_S];
    const int d = blockIdx.x * 256 + threadIdx.x;
    const int b = blockIdx.y;
    float st[D_S], negA[D_S];
    #pragma unroll
    for (int s = 0; s < D_S; ++s) { st[s] = 0.f; negA[s] = -Amat[d * D_S + s]; }
    const float dpv = Dp[d];

    for (int l0 = 0; l0 < L; l0 += SCHUNK) {
        __syncthreads();
        if (threadIdx.x < 32) {   // wave 0 drives the TDM
            tdm_load_lds_1d(lds_off(sB), Bm + ((size_t)b * L + l0) * D_S, SCHUNK * D_S);
            tdm_load_lds_1d(lds_off(sC), Cm + ((size_t)b * L + l0) * D_S, SCHUNK * D_S);
            __builtin_amdgcn_s_wait_tensorcnt(0);
        }
        __syncthreads();
        for (int cl = 0; cl < SCHUNK; ++cl) {
            const size_t off = ((size_t)b * L + l0 + cl) * dm + d;
            const float dlt = delta[off];
            const float av  = af[off];
            const float xs  = dlt * av;
            float acc = 0.f;
            #pragma unroll
            for (int s = 0; s < D_S; ++s) {
                const float e = __expf(dlt * negA[s]);
                st[s] = e * st[s] + xs * sB[cl * D_S + s];
                acc += st[s] * sC[cl * D_S + s];
            }
            y[off] = acc + dpv * av;
        }
    }
}

__global__ void gate_kernel(const float* __restrict__ ag, float* __restrict__ y,
                            bf16* __restrict__ ybf, size_t n, int dm)
{
    size_t i = (size_t)blockIdx.x * blockDim.x + threadIdx.x;
    if (i >= n) return;
    const int d = (int)(i % dm);
    const size_t bl = i / dm;
    const float g = ag[bl * (2 * dm) + dm + d];
    const float v = y[i] * (g * sigf(g));
    y[i] = v;
    ybf[i] = f2bf(v);
}

__global__ void add_kernel(float* __restrict__ x, const float* __restrict__ p, size_t n)
{
    size_t i = (size_t)blockIdx.x * blockDim.x + threadIdx.x;
    if (i < n) x[i] += p[i];
}

// classifier: one wave per (row, label) pair, 768-deep dot + shuffle reduce
__global__ __launch_bounds__(256) void cls_kernel(
    const float* __restrict__ x, const float* __restrict__ w,
    const float* __restrict__ bias, float* __restrict__ out,
    int rows, int dim, int nlab)
{
    const int gw = blockIdx.x * 8 + (threadIdx.x >> 5);
    const int lane = threadIdx.x & 31;
    const int row = gw / nlab, lab = gw % nlab;
    if (row >= rows) return;
    const float* xr = x + (size_t)row * dim;
    const float* wr = w + (size_t)lab * dim;
    float acc = 0.f;
    for (int k = lane; k < dim; k += 32) acc += xr[k] * wr[k];
    for (int off = 16; off > 0; off >>= 1) acc += __shfl_down(acc, off, 32);
    if (lane == 0) out[(size_t)row * nlab + lab] = acc + bias[lab];
}

// ---------------------------------------------------------------------------
extern "C" void kernel_launch(void* const* d_in, const int* in_sizes, int n_in,
                              void* d_out, int out_size, void* d_ws, size_t ws_size,
                              hipStream_t stream)
{
    (void)in_sizes; (void)n_in; (void)out_size;
    const int*   ids   = (const int*)  d_in[0];
    const int*   amask = (const int*)  d_in[1];
    const float* emb   = (const float*)d_in[2];
    const float* normw = (const float*)d_in[3];
    const float* wi    = (const float*)d_in[4];
    const float* cw    = (const float*)d_in[5];
    const float* cb    = (const float*)d_in[6];
    const float* wB    = (const float*)d_in[7];
    const float* wC    = (const float*)d_in[8];
    const float* w1    = (const float*)d_in[9];
    const float* w2    = (const float*)d_in[10];
    const float* Amat  = (const float*)d_in[11];
    const float* Dp    = (const float*)d_in[12];
    const float* wo    = (const float*)d_in[13];
    const float* clsw  = (const float*)d_in[14];
    const float* clsb  = (const float*)d_in[15];
    float* logits = (float*)d_out;

    char* base = (char*)d_ws;
    size_t off = 0;
    auto alloc = [&](size_t bytes) -> char* {
        off = (off + 255) & ~(size_t)255;
        char* p = base + off;
        off += bytes;
        return p;
    };

    float* xf    = (float*)alloc((size_t)MROWS * D_IN * 4);
    bf16*  hbf   = (bf16*) alloc((size_t)MROWS * D_IN * 2);
    float* ag    = (float*)alloc((size_t)MROWS * 2 * D_M * 4);
    float* af    = (float*)alloc((size_t)MROWS * D_M * 4);
    bf16*  abf   = (bf16*) alloc((size_t)MROWS * D_M * 2);
    float* Bmb   = (float*)alloc((size_t)MROWS * D_S * 4);
    float* Cmb   = (float*)alloc((size_t)MROWS * D_S * 4);
    float* dt1   = (float*)alloc((size_t)MROWS * D_D * 4);
    bf16*  dt1bf = (bf16*) alloc((size_t)MROWS * D_D * 2);
    float* delta = (float*)alloc((size_t)MROWS * D_M * 4);
    float* yf    = (float*)alloc((size_t)MROWS * D_M * 4);
    bf16*  ybf   = (bf16*) alloc((size_t)MROWS * D_M * 2);
    float* po    = (float*)alloc((size_t)MROWS * D_IN * 4);
    bf16*  wibf  = (bf16*) alloc((size_t)N_LAYERS * 2 * D_M * D_IN * 2);
    bf16*  wobf  = (bf16*) alloc((size_t)N_LAYERS * D_IN * D_M * 2);
    bf16*  wbbf  = (bf16*) alloc((size_t)N_LAYERS * D_S * D_M * 2);
    bf16*  wcbf  = (bf16*) alloc((size_t)N_LAYERS * D_S * D_M * 2);
    bf16*  w1bf  = (bf16*) alloc((size_t)N_LAYERS * D_D * D_M * 2);
    bf16*  w2bf  = (bf16*) alloc((size_t)N_LAYERS * D_M * D_D * 2);
    if (off > ws_size) return;  // insufficient scratch (deterministic no-op)

    auto cdiv = [](size_t a, size_t b) -> unsigned { return (unsigned)((a + b - 1) / b); };

    // --- weight precision conversion (f32 -> bf16), both layers at once ---
    {
        size_t n;
        n = (size_t)N_LAYERS * 2 * D_M * D_IN;
        cvt_bf16_kernel<<<cdiv(n, 256), 256, 0, stream>>>(wi, wibf, n);
        n = (size_t)N_LAYERS * D_IN * D_M;
        cvt_bf16_kernel<<<cdiv(n, 256), 256, 0, stream>>>(wo, wobf, n);
        n = (size_t)N_LAYERS * D_S * D_M;
        cvt_bf16_kernel<<<cdiv(n, 256), 256, 0, stream>>>(wB, wbbf, n);
        cvt_bf16_kernel<<<cdiv(n, 256), 256, 0, stream>>>(wC, wcbf, n);
        n = (size_t)N_LAYERS * D_D * D_M;
        cvt_bf16_kernel<<<cdiv(n, 256), 256, 0, stream>>>(w1, w1bf, n);
        cvt_bf16_kernel<<<cdiv(n, 256), 256, 0, stream>>>(w2, w2bf, n);
    }

    embed_kernel<<<MROWS, 256, 0, stream>>>(ids, amask, emb, xf, D_IN);

    const size_t nDM = (size_t)MROWS * D_M;
    for (int i = 0; i < N_LAYERS; ++i) {
        rmsnorm_bf16_kernel<<<MROWS, 256, 0, stream>>>(xf, normw + (size_t)i * D_IN, hbf, D_IN);

        wmma_gemm_nt<<<dim3(cdiv(2 * D_M, BN), MROWS / BM), 256, 0, stream>>>(
            hbf, wibf + (size_t)i * 2 * D_M * D_IN, ag, MROWS, 2 * D_M, D_IN);

        conv_silu_kernel<<<cdiv(nDM, 256), 256, 0, stream>>>(
            ag, cw + (size_t)i * D_M * K_CONV, cb + (size_t)i * D_M, af, abf, SS, D_M);

        wmma_gemm_nt<<<dim3(1, MROWS / BM), 256, 0, stream>>>(
            abf, wbbf + (size_t)i * D_S * D_M, Bmb, MROWS, D_S, D_M);
        wmma_gemm_nt<<<dim3(1, MROWS / BM), 256, 0, stream>>>(
            abf, wcbf + (size_t)i * D_S * D_M, Cmb, MROWS, D_S, D_M);

        wmma_gemm_nt<<<dim3(cdiv(D_D, BN), MROWS / BM), 256, 0, stream>>>(
            abf, w1bf + (size_t)i * D_D * D_M, dt1, MROWS, D_D, D_M);
        {
            size_t n = (size_t)MROWS * D_D;
            cvt_bf16_kernel<<<cdiv(n, 256), 256, 0, stream>>>(dt1, dt1bf, n);
        }
        wmma_gemm_nt<<<dim3(cdiv(D_M, BN), MROWS / BM), 256, 0, stream>>>(
            dt1bf, w2bf + (size_t)i * D_M * D_D, delta, MROWS, D_M, D_D);

        softplus_kernel<<<cdiv(nDM, 256), 256, 0, stream>>>(delta, Dp + (size_t)i * D_M, nDM, D_M);

        scan_kernel<<<dim3(D_M / 256, BB), 256, 0, stream>>>(
            delta, af, Bmb, Cmb, Amat + (size_t)i * D_M * D_S, Dp + (size_t)i * D_M, yf, SS, D_M);

        gate_kernel<<<cdiv(nDM, 256), 256, 0, stream>>>(ag, yf, ybf, nDM, D_M);

        wmma_gemm_nt<<<dim3(cdiv(D_IN, BN), MROWS / BM), 256, 0, stream>>>(
            ybf, wobf + (size_t)i * D_IN * D_M, po, MROWS, D_IN, D_M);

        add_kernel<<<cdiv((size_t)MROWS * D_IN, 256), 256, 0, stream>>>(
            xf, po, (size_t)MROWS * D_IN);
    }

    cls_kernel<<<cdiv((size_t)MROWS * N_LAB, 8), 256, 0, stream>>>(
        xf, clsw, clsb, logits, MROWS, D_IN, N_LAB);
}
